// SNN_50680614093259
// MI455X (gfx1250) — matrix operand order repaired and checked by reading, not compiled, not run on hardware
//
#include <hip/hip_runtime.h>
#include <stddef.h>
#include <stdint.h>

// ---------------------------------------------------------------------------
// Types
// ---------------------------------------------------------------------------
typedef __bf16 bf16;
typedef __attribute__((ext_vector_type(16))) __bf16 v16bf;
typedef __attribute__((ext_vector_type(8)))  __bf16 v8bf;
typedef __attribute__((ext_vector_type(8)))  float  v8f;

#define BN_SCALE 0.099999500003749977f   // 0.1 / sqrt(1 + 1e-5)

// Problem dims
#define BB   128
#define TT   100
#define DIN  700
#define DINP 704      // padded to multiple of 32
#define HH   1024
#define OO   20
#define N3P  128      // layer-3 N padded to one 128-row tile

#define LDSP 40       // LDS row stride in bf16 (80 B -> 16B-aligned fragments)
#define TILE (128 * LDSP)   // one staged [128 x 32] bf16 tile (padded)

__device__ __forceinline__ bf16 to_bf16(float f) {
    union { float f; unsigned u; } x; x.f = f;
    unsigned r = (x.u + 0x7FFFu + ((x.u >> 16) & 1u)) >> 16;
    union { unsigned short s; bf16 b; } y; y.s = (unsigned short)r;
    return y.b;
}

// Async global->LDS 16-byte copy (CDNA5 async path, ASYNCcnt-tracked).
// LDS byte address = low 32 bits of the flat shared pointer (ISA §10.2).
__device__ __forceinline__ void async_copy_b128(const bf16* g, bf16* l) {
    unsigned lds = (unsigned)(uintptr_t)l;
    asm volatile("global_load_async_to_lds_b128 %0, %1, off"
                 :: "v"(lds), "v"((unsigned long long)(uintptr_t)g)
                 : "memory");
}
__device__ __forceinline__ void wait_asynccnt0() {
    asm volatile("s_wait_asynccnt 0x0" ::: "memory");
}

// ---------------------------------------------------------------------------
// WMMA GEMM, transposed formulation:  computes C[b][n] = alpha*(X @ W^T) + bias
// by evaluating C^T = W @ X^T per 16x16 tile:
//   W  : [N, ldk] bf16 row-major (contiguous K)  -> WMMA A-operand
//   X  : row-major concat( X0[128 x K0], X1[128 x K1] ) bf16 -> WMMA B-operand
// Both operands have contiguous-K rows, so LDS staging is pure async b128
// copies (no transpose) and fragments are pure ds_load_b128 pairs.
// Double-buffered LDS: async-stage chunk kc+32 while WMMAs consume chunk kc.
// LDS sub-buffers: [0]=W ping, [1]=W pong, [2]=X ping, [3]=X pong (one array,
// runtime offsets only -- avoids addrspacecast constant initializers).
// Block: 256 threads (8 waves). Block tile: 128 N-rows x 128 batch.
// K0, K1 % 32 == 0. grid.x = N/128.
// ---------------------------------------------------------------------------
__global__ __launch_bounds__(256)
void snn_gemm_bf16_wmma(const bf16* __restrict__ X0, int lda0, int K0,
                        const bf16* __restrict__ X1, int lda1, int K1,
                        const bf16* __restrict__ W, int ldk,
                        const float* __restrict__ bias, float alpha,
                        float* __restrict__ Cf, int ldc,
                        bf16* __restrict__ Cb, int ldcb)
{
    __shared__ bf16 lds[4 * TILE];   // 40 KB of the 320 KB WGP pool

    const int tid  = threadIdx.x;
    const int wave = tid >> 5;
    const int lane = tid & 31;
    const int n0   = blockIdx.x * 128;
    const int m0   = wave * 16;          // wave's 16 output rows within block
    const int K    = K0 + K1;

    const v8f vzero = {0.f,0.f,0.f,0.f,0.f,0.f,0.f,0.f};
    v8f acc[8];
#pragma unroll
    for (int i = 0; i < 8; ++i) acc[i] = vzero;

    union Frag { v16bf v; struct Halves { v8bf lo, hi; } s; };

    // stage one K-chunk (32 wide) of X and W into LDS sub-buffer `buf`
    auto stage = [&](int buf, int kc) {
        bf16* dW = lds + (size_t)buf * TILE;
        bf16* dX = lds + (size_t)(2 + buf) * TILE;
#pragma unroll
        for (int it = 0; it < 2; ++it) {
            int task = tid + 256 * it;        // 512 tasks = 128 rows * 4 segs
            int row  = task >> 2;
            int seg  = task & 3;
            // X side (concat split; chunks never straddle the K0 boundary)
            int kk = kc + seg * 8;
            const bf16* src;
            int kl;
            if (kk < K0) { src = X0 + (size_t)row * lda0; kl = kk; }
            else         { src = X1 + (size_t)row * lda1; kl = kk - K0; }
            const bf16* gx = src + kl;
            __builtin_prefetch(gx + 64, 0, 3);           // 2 chunks ahead, RT/near
            async_copy_b128(gx, &dX[row * LDSP + seg * 8]);
            // W side
            const bf16* gw = W + (size_t)(n0 + row) * ldk + kc + seg * 8;
            __builtin_prefetch(gw + 64, 0, 3);
            async_copy_b128(gw, &dW[row * LDSP + seg * 8]);
        }
    };

    stage(0, 0);
    wait_asynccnt0();
    __syncthreads();

    int cur = 0;
    for (int kc = 0; kc < K; kc += 32) {
        // prefetch next chunk into the other buffer while we compute
        if (kc + 32 < K) stage(cur ^ 1, kc + 32);

        const bf16* sW = lds + (size_t)cur * TILE;
        const bf16* sX = lds + (size_t)(2 + cur) * TILE;

        // ---- A fragment: wave's 16 W-rows x 32 k ----
        // lane l: row = m0 + (l&15); l<16 -> K {0..7,16..23}, l>=16 -> {8..15,24..31}
        const int lrow = lane & 15;
        const int ah   = (lane >> 4) ? 8 : 0;
        const bf16* ap = &sW[(m0 + lrow) * LDSP + ah];
        Frag af;
        af.s.lo = *(const v8bf*)(ap);
        af.s.hi = *(const v8bf*)(ap + 16);

        // ---- preload all 8 X fragments (32 k x 16 b each), then WMMA chain --
        // lane l: col b = bt*16 + (l&15); l<16 -> K {0..7,16..23}, l>=16 -> {8..15,24..31}
        const int bh = (lane >> 4) ? 8 : 0;
        Frag xf[8];
#pragma unroll
        for (int bt = 0; bt < 8; ++bt) {
            const bf16* bp = &sX[(bt * 16 + (lane & 15)) * LDSP + bh];
            xf[bt].s.lo = *(const v8bf*)(bp);
            xf[bt].s.hi = *(const v8bf*)(bp + 16);
        }
#pragma unroll
        for (int bt = 0; bt < 8; ++bt) {
            acc[bt] = __builtin_amdgcn_wmma_f32_16x16x32_bf16(
                false, af.v, false, xf[bt].v, (short)0, acc[bt], false, false);
        }

        // next chunk's async copies must have landed; all waves done reading
        wait_asynccnt0();
        __syncthreads();
        cur ^= 1;
    }

    // ---- epilogue: D^T tile => element r, lane l:
    //        n = n0 + m0 + r + (l>>4)*8,  b = bt*16 + (l&15)
    const int rbase = (lane >> 4) * 8;
    const int lb    = lane & 15;
#pragma unroll
    for (int r = 0; r < 8; ++r) {
        int n = n0 + m0 + r + rbase;
        float bv = bias[n];
#pragma unroll
        for (int bt = 0; bt < 8; ++bt) {
            int b = bt * 16 + lb;
            float v = alpha * acc[bt][r] + bv;
            Cf[(size_t)b * ldc + n] = v;
            if (Cb) Cb[(size_t)b * ldcb + n] = to_bf16(v);
        }
    }
}

// ---------------------------------------------------------------------------
// Elementwise adaptive-LIF update for one hidden layer ([B,H] = 131072 elems)
// ---------------------------------------------------------------------------
__global__ void snn_update_kernel(const float* __restrict__ dense,
                                  const float* __restrict__ gm,
                                  const float* __restrict__ ga,
                                  float* __restrict__ mem_f,
                                  float* __restrict__ spk_f,
                                  float* __restrict__ bb_f,
                                  bf16*  __restrict__ mem_bf,
                                  bf16*  __restrict__ bb_bf,
                                  bf16*  __restrict__ spk_new_bf,
                                  bf16*  __restrict__ spk_old_bf,
                                  int n)
{
    int i = blockIdx.x * blockDim.x + threadIdx.x;
    if (i >= n) return;
    float tauM = 1.f / (1.f + __expf(-gm[i]));
    float tauA = 1.f / (1.f + __expf(-ga[i]));
    float spk_old = spk_f[i];
    float b   = tauA * bb_f[i] + (1.f - tauA) * spk_old;
    float Bth = 0.1f + 1.8f * b;
    float mem = mem_f[i];
    mem = mem + (dense[i] - mem) * tauM;
    float spk = (mem - Bth) > 0.f ? 1.f : 0.f;
    mem = (1.f - spk) * mem;
    mem_f[i] = mem; spk_f[i] = spk; bb_f[i] = b;
    mem_bf[i] = to_bf16(mem);
    bb_bf[i]  = to_bf16(b);
    spk_new_bf[i] = to_bf16(spk);
    if (spk_old_bf) spk_old_bf[i] = to_bf16(spk_old);
}

// ---------------------------------------------------------------------------
// Layer 3: tauM3 GEMV (K=40, N=20) + mem update + log_softmax over 20 classes
// ---------------------------------------------------------------------------
__global__ void snn_layer3_kernel(const float* __restrict__ dense3, int ld3,
                                  const float* __restrict__ w3m,
                                  const float* __restrict__ b3m,
                                  float* __restrict__ mem3_f,
                                  float* __restrict__ out_t)
{
    __shared__ float sd[OO], sm[OO], snm[OO];
    __shared__ float slse;
    int row = blockIdx.x, t = threadIdx.x;
    if (t < OO) {
        sd[t] = dense3[(size_t)row * ld3 + t];
        sm[t] = mem3_f[row * OO + t];
    }
    __syncthreads();
    float nm = 0.f;
    if (t < OO) {
        float g = b3m[t];
        for (int k = 0; k < OO; ++k) g += sd[k] * w3m[t * (2 * OO) + k];
        for (int k = 0; k < OO; ++k) g += sm[k] * w3m[t * (2 * OO) + OO + k];
        float tau = 1.f / (1.f + __expf(-g));
        nm = (1.f - tau) * sm[t] + sd[t] * tau;
        snm[t] = nm;
        mem3_f[row * OO + t] = nm;
    }
    __syncthreads();
    if (t == 0) {
        float mx = snm[0];
        for (int k = 1; k < OO; ++k) mx = fmaxf(mx, snm[k]);
        float s = 0.f;
        for (int k = 0; k < OO; ++k) s += __expf(snm[k] - mx);
        slse = mx + __logf(s);
    }
    __syncthreads();
    if (t < OO) out_t[row * OO + t] = nm - slse;
}

// ---------------------------------------------------------------------------
// One-time preprocessing kernels
// ---------------------------------------------------------------------------
// W [N,K] f32 row-major -> out rows [Np, ldk] bf16 at column offset koff,
// chunk padded to Kp columns; rows n>=N and cols k>=K are zero-filled.
__global__ void snn_convert_w_kernel(const float* __restrict__ W, int N, int K,
                                     bf16* __restrict__ out, int ldk, int koff,
                                     int Kp, int Np)
{
    int idx = blockIdx.x * blockDim.x + threadIdx.x;
    int total = Np * Kp;
    if (idx >= total) return;
    int n = idx / Kp, k = idx - n * Kp;
    float v = (n < N && k < K) ? W[(size_t)n * K + k] : 0.f;
    out[(size_t)n * ldk + koff + k] = to_bf16(v);
}

// inputs [B,T,DIN] f32 -> xbf [T,B,DINP] bf16 (cols 700..703 zero)
__global__ void snn_pack_x_kernel(const float* __restrict__ x, bf16* __restrict__ xbf)
{
    int idx = blockIdx.x * blockDim.x + threadIdx.x;
    int total = TT * BB * DINP;
    if (idx >= total) return;
    int k  = idx % DINP;
    int tb = idx / DINP;
    int b  = tb % BB;
    int t  = tb / BB;
    float v = (k < DIN) ? x[((size_t)b * TT + t) * DIN + k] : 0.f;
    xbf[idx] = to_bf16(v);
}

__global__ void snn_bias_prep_kernel(const float* __restrict__ b1x, const float* __restrict__ b1r,
                                     const float* __restrict__ b2x, const float* __restrict__ b2r,
                                     const float* __restrict__ b3x,
                                     float* __restrict__ bias1, float* __restrict__ bias2,
                                     float* __restrict__ bias3)
{
    int i = blockIdx.x * blockDim.x + threadIdx.x;
    if (i < HH) {
        bias1[i] = BN_SCALE * (b1x[i] + b1r[i]);
        bias2[i] = BN_SCALE * (b2x[i] + b2r[i]);
    }
    if (i < N3P) bias3[i] = (i < OO) ? BN_SCALE * b3x[i] : 0.f;
}

__global__ void snn_init_state_kernel(const float* __restrict__ h,
                                      float* __restrict__ st_f,
                                      bf16* __restrict__ st_bf, int n)
{
    int i = blockIdx.x * blockDim.x + threadIdx.x;
    if (i >= n) return;
    float v = h[i];
    st_f[i] = v;
    if (st_bf) st_bf[i] = to_bf16(v);
}

__global__ void snn_copy_out_kernel(const float* __restrict__ src,
                                    float* __restrict__ dst, int n)
{
    int i = blockIdx.x * blockDim.x + threadIdx.x;
    if (i < n) dst[i] = src[i];
}

// ---------------------------------------------------------------------------
// Host orchestration
// ---------------------------------------------------------------------------
static inline void launch_gemm(hipStream_t s,
                               const bf16* X0, int lda0, int K0,
                               const bf16* X1, int lda1, int K1,
                               const bf16* W, int ldk,
                               const float* bias, float alpha,
                               float* Cf, int ldc, bf16* Cb, int ldcb, int N)
{
    dim3 g(N / 128, 1, 1), b(256, 1, 1);
    snn_gemm_bf16_wmma<<<g, b, 0, s>>>(X0, lda0, K0, X1, lda1, K1,
                                       W, ldk, bias, alpha, Cf, ldc, Cb, ldcb);
}

extern "C" void kernel_launch(void* const* d_in, const int* in_sizes, int n_in,
                              void* d_out, int out_size, void* d_ws, size_t ws_size,
                              hipStream_t stream)
{
    (void)in_sizes; (void)n_in; (void)out_size; (void)ws_size;
    const float* inp = (const float*)d_in[0];
    const float* h[7];
    for (int i = 0; i < 7; ++i) h[i] = (const float*)d_in[1 + i];
    const float* w1x = (const float*)d_in[8];
    const float* b1x = (const float*)d_in[9];
    const float* w1r = (const float*)d_in[10];
    const float* b1r = (const float*)d_in[11];
    const float* w1m = (const float*)d_in[12];
    const float* b1m = (const float*)d_in[13];
    const float* w1a = (const float*)d_in[14];
    const float* b1a = (const float*)d_in[15];
    const float* w2x = (const float*)d_in[16];
    const float* b2x = (const float*)d_in[17];
    const float* w2r = (const float*)d_in[18];
    const float* b2r = (const float*)d_in[19];
    const float* w2m = (const float*)d_in[20];
    const float* b2m = (const float*)d_in[21];
    const float* w2a = (const float*)d_in[22];
    const float* b2a = (const float*)d_in[23];
    const float* w3x = (const float*)d_in[24];
    const float* b3x = (const float*)d_in[25];
    const float* w3m = (const float*)d_in[26];
    const float* b3m = (const float*)d_in[27];
    float* dout = (float*)d_out;

    // ---- workspace carve-up ----
    char* base = (char*)d_ws;
    size_t off = 0;
    auto alloc = [&](size_t bytes) -> char* {
        off = (off + 255) & ~(size_t)255;
        char* p = base + off;
        off += bytes;
        return p;
    };
    const int NH = BB * HH;                 // 131072
    const int KD1 = DINP + HH;              // 1728
    bf16* xbf  = (bf16*)alloc((size_t)TT * BB * DINP * 2);
    bf16* wbD1 = (bf16*)alloc((size_t)HH * KD1 * 2);     // [1024][1728] = w1x|w1r
    bf16* wbM1 = (bf16*)alloc((size_t)HH * 2 * HH * 2);  // [1024][2048]
    bf16* wbA1 = (bf16*)alloc((size_t)HH * 2 * HH * 2);
    bf16* wbD2 = (bf16*)alloc((size_t)HH * 2 * HH * 2);  // w2x|w2r
    bf16* wbM2 = (bf16*)alloc((size_t)HH * 2 * HH * 2);
    bf16* wbA2 = (bf16*)alloc((size_t)HH * 2 * HH * 2);
    bf16* wbD3 = (bf16*)alloc((size_t)N3P * HH * 2);     // [128][1024], rows>=20 zero
    float* bias1 = (float*)alloc(HH * 4);
    float* bias2 = (float*)alloc(HH * 4);
    float* bias3 = (float*)alloc(N3P * 4);
    float* mem1_f = (float*)alloc(NH * 4);
    float* spk1_f = (float*)alloc(NH * 4);
    float* bb1_f  = (float*)alloc(NH * 4);
    float* mem2_f = (float*)alloc(NH * 4);
    float* spk2_f = (float*)alloc(NH * 4);
    float* bb2_f  = (float*)alloc(NH * 4);
    float* mem3_f = (float*)alloc(BB * OO * 4);
    bf16* mem1_bf = (bf16*)alloc(NH * 2);
    bf16* bb1_bf  = (bf16*)alloc(NH * 2);
    bf16* spk1_bf = (bf16*)alloc(NH * 2);
    bf16* spk1_old_bf = (bf16*)alloc(NH * 2);
    bf16* mem2_bf = (bf16*)alloc(NH * 2);
    bf16* bb2_bf  = (bf16*)alloc(NH * 2);
    bf16* spk2_bf = (bf16*)alloc(NH * 2);
    bf16* dense1_bf = (bf16*)alloc(NH * 2);
    bf16* dense2_bf = (bf16*)alloc(NH * 2);
    float* dense1_f = (float*)alloc(NH * 4);
    float* gm1_f    = (float*)alloc(NH * 4);
    float* ga1_f    = (float*)alloc(NH * 4);
    float* dense2_f = (float*)alloc(NH * 4);
    float* gm2_f    = (float*)alloc(NH * 4);
    float* ga2_f    = (float*)alloc(NH * 4);
    float* dense3_f = (float*)alloc((size_t)BB * N3P * 4);

    const int TPB = 256;
    auto grid1 = [&](long n) { return dim3((unsigned)((n + TPB - 1) / TPB), 1, 1); };

    // ---- one-time preprocessing (all plain converts, no transposes) ----
    snn_pack_x_kernel<<<grid1((long)TT * BB * DINP), TPB, 0, stream>>>(inp, xbf);
    snn_convert_w_kernel<<<grid1((long)HH * DINP), TPB, 0, stream>>>(w1x, HH, DIN,  wbD1, KD1, 0,    DINP, HH);
    snn_convert_w_kernel<<<grid1((long)HH * HH),   TPB, 0, stream>>>(w1r, HH, HH,   wbD1, KD1, DINP, HH,   HH);
    snn_convert_w_kernel<<<grid1((long)HH * 2*HH), TPB, 0, stream>>>(w1m, HH, 2*HH, wbM1, 2*HH, 0,   2*HH, HH);
    snn_convert_w_kernel<<<grid1((long)HH * 2*HH), TPB, 0, stream>>>(w1a, HH, 2*HH, wbA1, 2*HH, 0,   2*HH, HH);
    snn_convert_w_kernel<<<grid1((long)HH * HH),   TPB, 0, stream>>>(w2x, HH, HH,   wbD2, 2*HH, 0,   HH,   HH);
    snn_convert_w_kernel<<<grid1((long)HH * HH),   TPB, 0, stream>>>(w2r, HH, HH,   wbD2, 2*HH, HH,  HH,   HH);
    snn_convert_w_kernel<<<grid1((long)HH * 2*HH), TPB, 0, stream>>>(w2m, HH, 2*HH, wbM2, 2*HH, 0,   2*HH, HH);
    snn_convert_w_kernel<<<grid1((long)HH * 2*HH), TPB, 0, stream>>>(w2a, HH, 2*HH, wbA2, 2*HH, 0,   2*HH, HH);
    snn_convert_w_kernel<<<grid1((long)N3P * HH),  TPB, 0, stream>>>(w3x, OO, HH,   wbD3, HH,   0,   HH,   N3P);
    snn_bias_prep_kernel<<<grid1(HH), TPB, 0, stream>>>(b1x, b1r, b2x, b2r, b3x, bias1, bias2, bias3);
    snn_init_state_kernel<<<grid1(NH), TPB, 0, stream>>>(h[0], mem1_f, mem1_bf, NH);
    snn_init_state_kernel<<<grid1(NH), TPB, 0, stream>>>(h[1], spk1_f, spk1_bf, NH);
    snn_init_state_kernel<<<grid1(NH), TPB, 0, stream>>>(h[2], bb1_f,  bb1_bf,  NH);
    snn_init_state_kernel<<<grid1(NH), TPB, 0, stream>>>(h[3], mem2_f, mem2_bf, NH);
    snn_init_state_kernel<<<grid1(NH), TPB, 0, stream>>>(h[4], spk2_f, spk2_bf, NH);
    snn_init_state_kernel<<<grid1(NH), TPB, 0, stream>>>(h[5], bb2_f,  bb2_bf,  NH);
    snn_init_state_kernel<<<grid1(BB * OO), TPB, 0, stream>>>(h[6], mem3_f, (bf16*)nullptr, BB * OO);

    // ---- T sequential steps ----
    for (int t = 0; t < TT; ++t) {
        const bf16* xt = xbf + (size_t)t * BB * DINP;
        // dense1 = BN_SCALE*(x@w1x^T + spk1@w1r^T) + bias1
        launch_gemm(stream, xt, DINP, DINP, spk1_bf, HH, HH,
                    wbD1, KD1, bias1, BN_SCALE, dense1_f, HH, dense1_bf, HH, HH);
        // gm1 = [dense1|mem1] @ w1m^T + b1m   (pre-sigmoid)
        launch_gemm(stream, dense1_bf, HH, HH, mem1_bf, HH, HH,
                    wbM1, 2*HH, b1m, 1.f, gm1_f, HH, (bf16*)nullptr, 0, HH);
        // ga1 = [dense1|bb1] @ w1a^T + b1a
        launch_gemm(stream, dense1_bf, HH, HH, bb1_bf, HH, HH,
                    wbA1, 2*HH, b1a, 1.f, ga1_f, HH, (bf16*)nullptr, 0, HH);
        snn_update_kernel<<<grid1(NH), TPB, 0, stream>>>(
            dense1_f, gm1_f, ga1_f, mem1_f, spk1_f, bb1_f,
            mem1_bf, bb1_bf, spk1_bf, spk1_old_bf, NH);
        // dense2 = BN_SCALE*(nspk1@w2x^T + spk1_old@w2r^T) + bias2
        launch_gemm(stream, spk1_bf, HH, HH, spk1_old_bf, HH, HH,
                    wbD2, 2*HH, bias2, BN_SCALE, dense2_f, HH, dense2_bf, HH, HH);
        launch_gemm(stream, dense2_bf, HH, HH, mem2_bf, HH, HH,
                    wbM2, 2*HH, b2m, 1.f, gm2_f, HH, (bf16*)nullptr, 0, HH);
        launch_gemm(stream, dense2_bf, HH, HH, bb2_bf, HH, HH,
                    wbA2, 2*HH, b2a, 1.f, ga2_f, HH, (bf16*)nullptr, 0, HH);
        snn_update_kernel<<<grid1(NH), TPB, 0, stream>>>(
            dense2_f, gm2_f, ga2_f, mem2_f, spk2_f, bb2_f,
            mem2_bf, bb2_bf, spk2_bf, (bf16*)nullptr, NH);
        // dense3 = BN_SCALE*(nspk2@w3x^T) + bias3  (N padded to 128)
        launch_gemm(stream, spk2_bf, HH, HH, spk2_bf, 0, 0,
                    wbD3, HH, bias3, BN_SCALE, dense3_f, N3P, (bf16*)nullptr, 0, N3P);
        snn_layer3_kernel<<<dim3(BB, 1, 1), dim3(32, 1, 1), 0, stream>>>(
            dense3_f, N3P, w3m, b3m, mem3_f, dout + (size_t)t * BB * OO);
    }

    // ---- final hidden states into d_out tail ----
    float* tail = dout + (size_t)TT * BB * OO;
    snn_copy_out_kernel<<<grid1(NH), TPB, 0, stream>>>(mem1_f, tail + 0 * NH, NH);
    snn_copy_out_kernel<<<grid1(NH), TPB, 0, stream>>>(spk1_f, tail + 1 * NH, NH);
    snn_copy_out_kernel<<<grid1(NH), TPB, 0, stream>>>(bb1_f,  tail + 2 * NH, NH);
    snn_copy_out_kernel<<<grid1(NH), TPB, 0, stream>>>(mem2_f, tail + 3 * NH, NH);
    snn_copy_out_kernel<<<grid1(NH), TPB, 0, stream>>>(spk2_f, tail + 4 * NH, NH);
    snn_copy_out_kernel<<<grid1(NH), TPB, 0, stream>>>(bb2_f,  tail + 5 * NH, NH);
    snn_copy_out_kernel<<<grid1(BB * OO), TPB, 0, stream>>>(mem3_f, tail + 6 * NH, BB * OO);
}
